// ResiduesDecode_21251498181158
// MI455X (gfx1250) — compile-verified, matrix operand order checked by reading, not run on hardware
//
#include <hip/hip_runtime.h>
#include <stdint.h>

// ---------------------------------------------------------------------------
// ResiduesDecode on MI455X (gfx1250, wave32)
//   out[b, base[r]+n, d] = sum_v x_v[b,r,v,d] * W[seq[r],n,v] + pos_ca[b,r,d]
// Memory-bound (201 MB x_v stream @ 23.3 TB/s ~= 9us). Compute mapped to
// V_WMMA_F32_16X16X4_F32 (exact f32); B operand staged via TDM tensor load
// with LDS padding (1 dword per 128) to kill bank conflicts; A padded to
// stride 132 for the same reason.
// ---------------------------------------------------------------------------

#define NLET  20
#define MAXN  13
#define B_    64
#define R_    2048
#define VDIM_ 128

__constant__ int g_natoms[NLET] = {4,10,7,7,5,8,8,3,9,7,7,8,7,10,6,5,6,13,11,6};

typedef float  v2f  __attribute__((ext_vector_type(2)));
typedef float  v8f  __attribute__((ext_vector_type(8)));
typedef unsigned int v4u __attribute__((ext_vector_type(4)));
typedef int    v4i  __attribute__((ext_vector_type(4)));
typedef int    v8i  __attribute__((ext_vector_type(8)));

#if __has_builtin(__builtin_amdgcn_wmma_f32_16x16x4_f32)
#define USE_WMMA_F32 1
#else
#define USE_WMMA_F32 0
#endif
#if !USE_WMMA_F32 && __has_builtin(__builtin_amdgcn_wmma_f32_16x16x32_bf16)
#define USE_WMMA_BF16 1
typedef __bf16 v16bf __attribute__((ext_vector_type(16)));
#else
#define USE_WMMA_BF16 0
#endif
#if __has_builtin(__builtin_amdgcn_tensor_load_to_lds) && __has_builtin(__builtin_amdgcn_s_wait_tensorcnt)
#define USE_TDM 1
#else
#define USE_TDM 0
#endif

// ---------------------------------------------------------------------------
// Kernel 0: exclusive prefix sum of NATOMS[seq[r]] -> atom base offset per r
// ---------------------------------------------------------------------------
__global__ __launch_bounds__(256)
void atom_base_scan_kernel(const int* __restrict__ seq, int* __restrict__ base) {
  __shared__ int part[256];
  const int t  = threadIdx.x;
  const int CH = R_ / 256; // 8 residues per thread
  int loc[CH];
  int s = 0;
  for (int i = 0; i < CH; ++i) { loc[i] = s; s += g_natoms[seq[t * CH + i]]; }
  part[t] = s;
  __syncthreads();
  if (t == 0) {
    int acc = 0;
    for (int i = 0; i < 256; ++i) { int v = part[i]; part[i] = acc; acc += v; }
  }
  __syncthreads();
  const int off = part[t];
  for (int i = 0; i < CH; ++i) base[t * CH + i] = off + loc[i];
}

// ---------------------------------------------------------------------------
// Kernel 1: one workgroup per residue. 384 threads = 12 waves.
// Per-residue GEMM: A = W[seq[r]] (16x128, rows>=13 zero),
//                   B[v, col] = x_v[b,r,v,d], col = b*3+d  (128x192),
// wave w computes 16x16 output tile w (N tiles = 192/16 = 12).
//
// LDS layout (single block at LDS offset 0, required by the TDM descriptor):
//   B: rows of 384 f32 + 1 f32 pad per 128 -> stride 387; elem e of row b at
//      smem[b*387 + e + (e>>7)]    (matches TDM pad_interval=128dw/amount=1dw)
//   A: 16 rows, padded stride 132  (banks 4m apart -> conflict-free b64 loads)
//   P: pos_ca slice, col-indexed (b*3+d)
// ---------------------------------------------------------------------------
#define B_STRIDE 387
#define A_STRIDE 132
#define LDSB_F   (B_ * B_STRIDE)            // 24768
#define LDSA_F   (16 * A_STRIDE)            // 2112
#define LDSP_F   192
#define LDSA_O   LDSB_F
#define LDSP_O   (LDSB_F + LDSA_F)
#define SMEM_F   (LDSB_F + LDSA_F + LDSP_F) // 27072 f32 = 108288 B

__global__ __launch_bounds__(384)
void residues_decode_kernel(const float* __restrict__ pos_ca,
                            const float* __restrict__ x_v,
                            const float* __restrict__ W,
                            const int*   __restrict__ seq,
                            const int*   __restrict__ base,
                            float*       __restrict__ out,
                            int total_atoms) {
  __shared__ __align__(16) float smem[SMEM_F];

  const int r         = blockIdx.x;
  const int tid       = threadIdx.x;
  const int seq_r     = seq[r];
  const int natom     = g_natoms[seq_r];
  const int atom_base = base[r];

#if USE_TDM
  // --- Stage B via Tensor Data Mover: 2-D tile, 64 rows x 1536B, row stride
  // R*1536B, into LDS offset 0 with 1-dword padding per 128 dwords.
  // Issued once by wave 0 (EXEC ignored by TDM).
  if (tid < 32) {
    const uint64_t ga = (uint64_t)(uintptr_t)(x_v + (size_t)r * (VDIM_ * 3));
    v4u g0;
    g0.x = 1u;                                            // count=1, user D#
    g0.y = 0u;                                            // lds_addr = 0
    g0.z = (unsigned)(ga & 0xffffffffu);                  // global_addr[31:0]
    g0.w = (unsigned)((ga >> 32) & 0x1ffffffu) | (2u << 30); // addr[56:32]|type=2
    v8i g1;
    g1[0] = (int)((2u << 16)      // data_size = 2 (4 bytes)
                | (1u << 20)      // pad_enable
                | (6u << 22)      // pad_interval: 2^(6+1) = 128 dwords
                | (0u << 25));    // pad_amount: 1 dword
    g1[1] = (int)(384u << 16);    // tensor_dim0 = 384 elements
    g1[2] = (int)(64u << 16);     // tensor_dim1 = 64 rows
    g1[3] = (int)(384u << 16);    // tile_dim0   = 384
    g1[4] = 64;                   // tile_dim1   = 64
    g1[5] = (int)(R_ * VDIM_ * 3);// tensor_dim0_stride = 786432 elements
    g1[6] = 0;
    g1[7] = 0;
    v4i g2 = {0, 0, 0, 0};
    v4i g3 = {0, 0, 0, 0};
    v8i g4 = {0, 0, 0, 0, 0, 0, 0, 0};  // unused trailing descriptor group
    __builtin_amdgcn_tensor_load_to_lds(g0, g1, g2, g3, g4, 0);
  }
#else
  // Fallback: cooperative copy of the x_v slice with the same padded mapping.
  for (int u = tid; u < 6144; u += 384) {
    const int b = u / 96, q4 = u % 96;
    const float4 v = *(const float4*)(x_v + ((size_t)b * R_ + r) * (VDIM_ * 3) + q4 * 4);
    const int e = q4 * 4;                          // 4 elems never straddle a pad
    float* dst = &smem[b * B_STRIDE + e + (e >> 7)];
    dst[0] = v.x; dst[1] = v.y; dst[2] = v.z; dst[3] = v.w;
  }
#endif

  // --- Stage A (W[seq_r], 13x128 -> 16 x A_STRIDE, zero-padded) and pos_ca.
  {
    const float* Wr = W + (size_t)seq_r * (MAXN * VDIM_);
    for (int i4 = tid; i4 < 512; i4 += 384) {
      const int i   = i4 * 4;
      const int row = i >> 7;
      const int kc  = i & 127;
      float4 v = make_float4(0.f, 0.f, 0.f, 0.f);
      if (row < MAXN) v = *(const float4*)(Wr + i);
      *(float4*)(&smem[LDSA_O + row * A_STRIDE + kc]) = v;
    }
    for (int i = tid; i < 192; i += 384) {
      const int b = i / 3, d = i % 3;
      smem[LDSP_O + i] = pos_ca[((size_t)b * R_ + r) * 3 + d];
    }
  }

#if USE_TDM
  if (tid < 32) __builtin_amdgcn_s_wait_tensorcnt(0);
#endif
  __syncthreads();

  // --- Compute. Lane layout per CDNA5 WMMA f32 16x16x4:
  //   A: lane%16 = M, lane/16 = K-half, VGPR j -> K = 2*half + j
  //   B: lane%16 = N, lane/16 = K-half, VGPR j -> K = 2*half + j
  //   C/D: N = lane%16, VGPR j -> M = j + 8*(lane/16)
  const int wave = tid >> 5;
  const int lane = tid & 31;
  const int half = lane >> 4;
  const int nlo  = lane & 15;
  const int col  = wave * 16 + nlo;   // 0..191 == b*3 + d
  const int bcol = col / 3;
  const int dcol = col % 3;
  const int bRow = bcol * B_STRIDE;

  v8f acc = {0.f, 0.f, 0.f, 0.f, 0.f, 0.f, 0.f, 0.f};

#if USE_WMMA_F32
  {
    const int aOff  = LDSA_O + nlo * A_STRIDE + 2 * half; // A[m, 2*half + k0]
    const int eBase = dcol + 6 * half;                    // B elem idx at k0=0
    #pragma unroll 4
    for (int k0 = 0; k0 < VDIM_; k0 += 4) {
      const int e0 = eBase + 3 * k0;       // element (k = k0+2*half,   col)
      const int e1 = e0 + 3;               // element (k = k0+2*half+1, col)
      v2f a, b;
      a.x = smem[aOff + k0];
      a.y = smem[aOff + k0 + 1];
      b.x = smem[bRow + e0 + (e0 >> 7)];   // pad-aware LDS index
      b.y = smem[bRow + e1 + (e1 >> 7)];
      acc = __builtin_amdgcn_wmma_f32_16x16x4_f32(false, a, false, b,
                                                  (short)0, acc, false, false);
    }
  }
#elif USE_WMMA_BF16
  {
    for (int k0 = 0; k0 < VDIM_; k0 += 32) {
      v16bf a, b;
      #pragma unroll
      for (int i = 0; i < 16; ++i) {
        const int kk = k0 + (i < 8 ? 8 * half + i : 16 + 8 * half + (i - 8));
        const int e  = kk * 3 + dcol;
        a[i] = (__bf16)smem[LDSA_O + nlo * A_STRIDE + kk];
        b[i] = (__bf16)smem[bRow + e + (e >> 7)];
      }
      acc = __builtin_amdgcn_wmma_f32_16x16x32_bf16(false, a, false, b,
                                                    (short)0, acc, false, false);
    }
  }
#else
  for (int k = 0; k < VDIM_; ++k) {
    const int e = k * 3 + dcol;
    const float bv = smem[bRow + e + (e >> 7)];
    #pragma unroll
    for (int j = 0; j < 8; ++j)
      acc[j] += smem[LDSA_O + (j + 8 * half) * A_STRIDE + k] * bv;
  }
#endif

  // --- Epilogue: add CA offset, write only valid atoms (ragged output).
  const float posv  = smem[LDSP_O + col];
  const size_t oRow = ((size_t)bcol * (size_t)total_atoms + (size_t)atom_base) * 3 + dcol;
  #pragma unroll
  for (int j = 0; j < 8; ++j) {
    const int M = j + 8 * half;
    if (M < natom) out[oRow + (size_t)M * 3] = acc[j] + posv;
  }
}

// ---------------------------------------------------------------------------
extern "C" void kernel_launch(void* const* d_in, const int* in_sizes, int n_in,
                              void* d_out, int out_size, void* d_ws, size_t ws_size,
                              hipStream_t stream) {
  const float* pos_ca = (const float*)d_in[0];
  const float* x_v    = (const float*)d_in[1];
  const float* W      = (const float*)d_in[2];
  const int*   seq    = (const int*)d_in[3];
  const int total_atoms = in_sizes[4];   // len(res_idx)

  int* base = (int*)d_ws;                // R_ ints of scratch

  atom_base_scan_kernel<<<1, 256, 0, stream>>>(seq, base);
  residues_decode_kernel<<<R_, 384, 0, stream>>>(pos_ca, x_v, W, seq, base,
                                                 (float*)d_out, total_atoms);
}
